// MRU_78039555768550
// MI455X (gfx1250) — matrix operand order repaired and checked by reading, not compiled, not run on hardware
//
#include <hip/hip_runtime.h>

#define B_  8
#define L_  2048
#define D_  256
#define H_  8
#define M_  (L_ * B_)     // 16384
#define HD_ (H_ * D_)     // 2048

typedef __attribute__((ext_vector_type(16))) __bf16 bf16x16;
typedef __attribute__((ext_vector_type(8)))  float  f32x8;

struct alignas(16) U4 { unsigned int a, b, c, d; };

union BFrag {
    bf16x16 v;
    U4      q[2];
};

union LDV {
    U4 q;
    unsigned short s[8];
};

__device__ __forceinline__ unsigned short f2bf(float f) {
    unsigned int u = __float_as_uint(f);
    u += 0x7fffu + ((u >> 16) & 1u);         // round-to-nearest-even
    return (unsigned short)(u >> 16);
}
__device__ __forceinline__ float bf2f(unsigned short h) {
    return __uint_as_float(((unsigned int)h) << 16);
}

// ---- WMMA fragment loaders (layouts per CDNA5 ISA 7.12.2) ----
// A 16x32 bf16: lane holds row = lane&15; VGPR0-3: K = half*8+[0..7], VGPR4-7: +16
__device__ __forceinline__ BFrag load_a(const unsigned short* base, int ldk, int kk, int lane) {
    int row = lane & 15;
    int k0  = kk * 32 + ((lane >> 4) << 3);
    BFrag f;
    const unsigned short* p = base + row * ldk + k0;
    f.q[0] = *(const U4*)p;
    f.q[1] = *(const U4*)(p + 16);
    return f;
}
// B 32x16 bf16 from B^T storage [N][K]: lane holds col = lane&15; VGPR0-3: K = half*8+[0..7], VGPR4-7: +16
__device__ __forceinline__ BFrag load_b(const unsigned short* base, int ldk, int kk, int lane) {
    int col = lane & 15;
    int k0  = kk * 32 + ((lane >> 4) << 3);
    BFrag f;
    const unsigned short* p = base + col * ldk + k0;
    f.q[0] = *(const U4*)p;
    f.q[1] = *(const U4*)(p + 16);
    return f;
}

__device__ __forceinline__ f32x8 wmma_bf16(const BFrag& a, const BFrag& b, f32x8 c) {
    return __builtin_amdgcn_wmma_f32_16x16x32_bf16(false, a.v, false, b.v, (short)0, c, false, false);
}

// ---------------- prep kernels (fp32 -> bf16, transposes) ----------------
__global__ void prep_x_kernel(const float* __restrict__ x, unsigned short* __restrict__ xr) {
    int idx = blockIdx.x * blockDim.x + threadIdx.x;
    const int total = B_ * L_ * D_;
    for (; idx < total; idx += gridDim.x * blockDim.x) {
        int d = idx & (D_ - 1);
        int t = (idx >> 8) & (L_ - 1);
        int b = idx >> 19;                       // idx / (L*D)
        xr[((t * B_ + b) << 8) + d] = f2bf(x[idx]);
    }
}

// W (H, 2D, D) -> WxT[h][e][d] = W[h][d][e], WhT[h][e][d] = W[h][D+d][e]
__global__ void prep_w_kernel(const float* __restrict__ W,
                              unsigned short* __restrict__ WxT,
                              unsigned short* __restrict__ WhT) {
    int idx = blockIdx.x * blockDim.x + threadIdx.x;
    const int total = H_ * 2 * D_ * D_;
    for (; idx < total; idx += gridDim.x * blockDim.x) {
        int e  = idx & (D_ - 1);
        int d2 = (idx >> 8) & (2 * D_ - 1);
        int h  = idx >> 17;
        unsigned short v = f2bf(W[idx]);
        if (d2 < D_) WxT[(h * D_ + e) * D_ + d2]        = v;
        else         WhT[(h * D_ + e) * D_ + (d2 - D_)] = v;
    }
}

// Wl (HD, D) -> WlT[e][m]
__global__ void prep_wl_kernel(const float* __restrict__ Wl, unsigned short* __restrict__ WlT) {
    int idx = blockIdx.x * blockDim.x + threadIdx.x;
    const int total = HD_ * D_;
    for (; idx < total; idx += gridDim.x * blockDim.x) {
        int e = idx & (D_ - 1);
        int m = idx >> 8;
        WlT[e * HD_ + m] = f2bf(Wl[idx]);
    }
}

// ---------------- input projections ----------------
// XF/XC layout: [h][t][col][b] (8 bf16 contiguous in b -> one b128 per lane in the scan)
// grid (1024 m-strips, H, 2), block 256 = 8 waves, each wave 2 n-tiles, K=256
__global__ void proj_kernel(const unsigned short* __restrict__ xr,
                            const unsigned short* __restrict__ WfxT,
                            const unsigned short* __restrict__ WcxT,
                            unsigned short* __restrict__ XF,
                            unsigned short* __restrict__ XC) {
    int ms   = blockIdx.x;
    int h    = blockIdx.y;
    int sel  = blockIdx.z;
    int lane = threadIdx.x & 31;
    int wave = threadIdx.x >> 5;
    const unsigned short* BT  = (sel ? WcxT : WfxT) + h * D_ * D_;
    unsigned short*       OUT = (sel ? XC : XF) + h * M_ * D_;
    const unsigned short* Abase = xr + (ms * 16) * D_;    // rows m = t*8 + b
    for (int half = 0; half < 2; ++half) {
        int nt = wave * 2 + half;
        const unsigned short* Bbase = BT + nt * 16 * D_;
        f32x8 acc = {};
        #pragma unroll
        for (int kk = 0; kk < 8; ++kk) {
            BFrag a = load_a(Abase, D_, kk, lane);
            BFrag b = load_b(Bbase, D_, kk, lane);
            acc = wmma_bf16(a, b, acc);
        }
        // C rows v map to m = ms*16 + (lane>=16 ? 8 : 0) + v -> t = ms*2 + (lane>>4), b = v
        int col = nt * 16 + (lane & 15);
        int t   = ms * 2 + (lane >> 4);
        LDV o;
        #pragma unroll
        for (int v = 0; v < 8; ++v) o.s[v] = f2bf(acc[v]);
        *(U4*)(OUT + (((size_t)t * D_ + col) << 3)) = o.q;
    }
}

// ---------------- recurrence: one workgroup per head ----------------
// block = 512 threads = 16 waves; wave w owns n-tile w of D=256.
// Weights live in VGPRs (128 regs of B-frags), h-state fp32 lives in the C-layout
// registers of each lane, LDS holds only the 16x256 bf16 A-tile for cross-wave exchange.
__global__ void recur_kernel(const unsigned short* __restrict__ XF,
                             const unsigned short* __restrict__ XC,
                             const unsigned short* __restrict__ WfhT,
                             const unsigned short* __restrict__ WchT,
                             unsigned short* __restrict__ HS) {
    __shared__ unsigned short sA[16 * D_];                // 8 KB state A-tile (rows 8-15 = pad)

    int h    = blockIdx.x;
    int tid  = threadIdx.x;
    int lane = tid & 31;
    int wave = tid >> 5;                                  // = n-tile

    // zero the state tile (512 threads * 16B = 8KB)
    U4 z = {0u, 0u, 0u, 0u};
    ((U4*)sA)[tid] = z;

    // preload this wave's recurrent-weight B fragments into registers
    const unsigned short* gBf = WfhT + h * D_ * D_ + wave * 16 * D_;
    const unsigned short* gBc = WchT + h * D_ * D_ + wave * 16 * D_;
    BFrag bf[8], bc[8];
    #pragma unroll
    for (int kk = 0; kk < 8; ++kk) {
        bf[kk] = load_b(gBf, D_, kk, lane);
        bc[kk] = load_b(gBc, D_, kk, lane);
    }

    int col = wave * 16 + (lane & 15);
    const U4* xfp = (const U4*)(XF + (size_t)h * M_ * D_);   // [t][col] -> U4 of 8 batches
    const U4* xcp = (const U4*)(XC + (size_t)h * M_ * D_);
    unsigned short* hsbase = HS + h * D_ + col;

    float hreg[8];
    #pragma unroll
    for (int v = 0; v < 8; ++v) hreg[v] = 0.0f;

    __syncthreads();

    for (int t = 0; t < L_; ++t) {
        // issue gate-input loads first; latency hides under the WMMA chain
        LDV xfv, xcv;
        xfv.q = xfp[t * D_ + col];
        xcv.q = xcp[t * D_ + col];

        f32x8 facc = {}, cacc = {};
        #pragma unroll
        for (int kk = 0; kk < 8; ++kk) {                  // K = 256
            BFrag a = load_a(sA, D_, kk, lane);
            facc = wmma_bf16(a, bf[kk], facc);
            cacc = wmma_bf16(a, bc[kk], cacc);
        }
        __syncthreads();                                  // all reads of sA done before update
        if (lane < 16) {                                  // rows 0-7 (real batch) in lanes 0-15
            #pragma unroll
            for (int v = 0; v < 8; ++v) {
                float xf = bf2f(xfv.s[v]);
                float xc = bf2f(xcv.s[v]);
                float fg = __builtin_amdgcn_rcpf(1.0f + __expf(-(facc[v] + xf)));
                float e2 = __expf(2.0f * (cacc[v] + xc));
                float ng = 1.0f - 2.0f * __builtin_amdgcn_rcpf(e2 + 1.0f);
                float hnew = fg * hreg[v] + (1.0f - fg) * ng;
                hreg[v] = hnew;
                unsigned short hb = f2bf(hnew);
                sA[v * D_ + col] = hb;
                hsbase[(size_t)(v * L_ + t) * HD_] = hb;  // HS row m = b*L + t
            }
        }
        __syncthreads();                                  // sA updated before next step's reads
    }
}

// ---------------- output projection: out = HS(16384x2048) @ Wl + bl ----------------
__global__ void final_kernel(const unsigned short* __restrict__ HS,
                             const unsigned short* __restrict__ WlT,
                             const float* __restrict__ bl,
                             float* __restrict__ out) {
    int ms   = blockIdx.x;
    int lane = threadIdx.x & 31;
    int wave = threadIdx.x >> 5;
    const unsigned short* Abase = HS + (size_t)(ms * 16) * HD_;
    for (int half = 0; half < 2; ++half) {
        int nt = wave * 2 + half;
        const unsigned short* Bbase = WlT + nt * 16 * HD_;
        f32x8 acc = {};
        for (int kk = 0; kk < 64; ++kk) {                 // K = 2048
            BFrag a = load_a(Abase, HD_, kk, lane);
            BFrag b = load_b(Bbase, HD_, kk, lane);
            acc = wmma_bf16(a, b, acc);
        }
        int col    = nt * 16 + (lane & 15);
        float bias = bl[col];
        int rbase  = ms * 16 + ((lane >> 4) << 3);
        #pragma unroll
        for (int v = 0; v < 8; ++v)
            out[(rbase + v) * D_ + col] = acc[v] + bias;
    }
}

extern "C" void kernel_launch(void* const* d_in, const int* in_sizes, int n_in,
                              void* d_out, int out_size, void* d_ws, size_t ws_size,
                              hipStream_t stream) {
    (void)in_sizes; (void)n_in; (void)out_size; (void)ws_size;
    const float* x  = (const float*)d_in[0];
    const float* Wf = (const float*)d_in[1];
    const float* Wc = (const float*)d_in[2];
    const float* Wl = (const float*)d_in[3];
    const float* bl = (const float*)d_in[4];
    float* out = (float*)d_out;

    char* ws = (char*)d_ws;
    size_t off = 0;
    auto carve = [&](size_t bytes) { char* p = ws + off; off += (bytes + 255) & ~(size_t)255; return p; };
    unsigned short* xr   = (unsigned short*)carve((size_t)M_ * D_ * 2);        //   8 MB
    unsigned short* WfxT = (unsigned short*)carve((size_t)H_ * D_ * D_ * 2);   //   1 MB
    unsigned short* WfhT = (unsigned short*)carve((size_t)H_ * D_ * D_ * 2);
    unsigned short* WcxT = (unsigned short*)carve((size_t)H_ * D_ * D_ * 2);
    unsigned short* WchT = (unsigned short*)carve((size_t)H_ * D_ * D_ * 2);
    unsigned short* WlT  = (unsigned short*)carve((size_t)D_ * HD_ * 2);       //   1 MB
    unsigned short* XF   = (unsigned short*)carve((size_t)H_ * M_ * D_ * 2);   //  64 MB
    unsigned short* XC   = (unsigned short*)carve((size_t)H_ * M_ * D_ * 2);   //  64 MB
    unsigned short* HS   = (unsigned short*)carve((size_t)M_ * HD_ * 2);       //  64 MB

    prep_x_kernel <<<4096, 256, 0, stream>>>(x, xr);
    prep_w_kernel <<<2048, 256, 0, stream>>>(Wf, WfxT, WfhT);
    prep_w_kernel <<<2048, 256, 0, stream>>>(Wc, WcxT, WchT);
    prep_wl_kernel<<< 512, 256, 0, stream>>>(Wl, WlT);
    proj_kernel   <<<dim3(M_ / 16, H_, 2), 256, 0, stream>>>(xr, WfxT, WcxT, XF, XC);
    recur_kernel  <<<H_, 512, 0, stream>>>(XF, XC, WfhT, WchT, HS);
    final_kernel  <<<M_ / 16, 256, 0, stream>>>(HS, WlT, bl, out);
}
